// MLA_28363964023124
// MI455X (gfx1250) — matrix operand order repaired and checked by reading, not compiled, not run on hardware
//
#include <hip/hip_runtime.h>
#include <hip/hip_bf16.h>
#include <cstdint>

// ---- problem dims (match reference) ----
#define B_     2
#define S_     2048
#define D_     2048
#define H_     16
#define QLORA  1536
#define KVLORA 512
#define DNOPE  128
#define DROPE  64
#define DV     128
#define QKD    192           // DNOPE + DROPE
#define R_     (B_ * S_)     // 4096 token rows

typedef __attribute__((ext_vector_type(16))) __bf16 v16bf;
typedef __attribute__((ext_vector_type(8)))  __bf16 v8bf;
typedef __attribute__((ext_vector_type(4)))  __bf16 v4bf;
typedef __attribute__((ext_vector_type(8)))  float  v8f;
typedef __attribute__((ext_vector_type(4)))  float  v4f;

static __device__ __forceinline__ v8f wmma_bf16(v16bf a, v16bf b, v8f c) {
  // v_wmma_f32_16x16x32_bf16 : D = A(16x32) * B(32x16) + C(16x16 f32)
  return __builtin_amdgcn_wmma_f32_16x16x32_bf16(false, a, false, b, (short)0, c,
                                                 false, false);
}

// A-matrix 16x32 bf16 fragment: lane L holds row M=L&15; K range
// {hi*16 .. hi*16+15} (hi = L>>4) -> two 16B loads.
static __device__ __forceinline__ v16bf load_a_frag(const __bf16* rowp) {
  v8bf a0 = *(const v8bf*)(rowp);
  v8bf a1 = *(const v8bf*)(rowp + 8);
  v16bf a;
#pragma unroll
  for (int i = 0; i < 8; ++i) { a[i] = a0[i]; a[i + 8] = a1[i]; }
  return a;
}

// ---------------- elementwise: fp32 -> bf16 (vectorized x4) ----------------
__global__ void k_f32_to_bf16(const float* __restrict__ in, __bf16* __restrict__ out,
                              size_t n4) {   // n4 = n/4
  size_t i = (size_t)blockIdx.x * blockDim.x + threadIdx.x;
  size_t stride = (size_t)gridDim.x * blockDim.x;
  for (; i < n4; i += stride) {
    v4f x = *(const v4f*)(in + 4 * i);
    v4bf y;
#pragma unroll
    for (int j = 0; j < 4; ++j) y[j] = (__bf16)x[j];
    *(v4bf*)(out + 4 * i) = y;
  }
}

// ---------------- RMS norm rows (fp32 in, bf16 out) ----------------
__global__ void k_rmsnorm(const float* __restrict__ in, const float* __restrict__ w,
                          __bf16* __restrict__ out, int width) {
  int row = blockIdx.x;
  const float* r = in + (size_t)row * width;
  float ss = 0.f;
  for (int i = threadIdx.x; i < width; i += blockDim.x) { float v = r[i]; ss += v * v; }
  __shared__ float red[256];
  red[threadIdx.x] = ss;
  __syncthreads();
  for (int s = 128; s > 0; s >>= 1) {
    if ((int)threadIdx.x < s) red[threadIdx.x] += red[threadIdx.x + s];
    __syncthreads();
  }
  float scl = rsqrtf(red[0] / (float)width + 1e-6f);
  for (int i = threadIdx.x; i < width; i += blockDim.x)
    out[(size_t)row * width + i] = (__bf16)(r[i] * scl * w[i]);
}

// ------- kv split: rmsnorm first 512 cols -> bf16; rope last 64 -> fp32 -------
__global__ void k_kv_split(const float* __restrict__ kvf, const float* __restrict__ w,
                           const float* __restrict__ fcos, const float* __restrict__ fsin,
                           __bf16* __restrict__ kvn, float* __restrict__ kpe) {
  int row = blockIdx.x;          // b*S + s
  int s = row % S_;
  const float* r = kvf + (size_t)row * (KVLORA + DROPE);
  float ss = 0.f;
  for (int i = threadIdx.x; i < KVLORA; i += blockDim.x) { float v = r[i]; ss += v * v; }
  __shared__ float red[256];
  red[threadIdx.x] = ss;
  __syncthreads();
  for (int t = 128; t > 0; t >>= 1) {
    if ((int)threadIdx.x < t) red[threadIdx.x] += red[threadIdx.x + t];
    __syncthreads();
  }
  float scl = rsqrtf(red[0] / (float)KVLORA + 1e-6f);
  for (int i = threadIdx.x; i < KVLORA; i += blockDim.x)
    kvn[(size_t)row * KVLORA + i] = (__bf16)(r[i] * scl * w[i]);
  if ((int)threadIdx.x < DROPE / 2) {
    int i = threadIdx.x;
    float xr = r[KVLORA + 2 * i], xi = r[KVLORA + 2 * i + 1];
    float c  = fcos[(size_t)s * (DROPE / 2) + i];
    float sn = fsin[(size_t)s * (DROPE / 2) + i];
    kpe[(size_t)row * DROPE + 2 * i]     = xr * c - xi * sn;
    kpe[(size_t)row * DROPE + 2 * i + 1] = xr * sn + xi * c;
  }
}

// ------- assemble per-(b,h) attention operands: Q(S,192) K(S,192) Vt(128,S) -------
__global__ void k_assemble(const float* __restrict__ q_f, const float* __restrict__ kvb,
                           const float* __restrict__ kpe,
                           const float* __restrict__ fcos, const float* __restrict__ fsin,
                           __bf16* __restrict__ Qh, __bf16* __restrict__ Kh,
                           __bf16* __restrict__ Vt) {
  int row = blockIdx.x;          // b*S + s
  int h   = blockIdx.y;
  int b = row / S_, s = row % S_;
  int d = threadIdx.x;           // 0..191
  size_t bh = (size_t)(b * H_ + h);
  const float* qrow = q_f + (size_t)row * (H_ * QKD) + (size_t)h * QKD;

  float qv;
  if (d < DNOPE) {
    qv = qrow[d];
  } else {
    int i = (d - DNOPE) >> 1;
    float xr = qrow[DNOPE + 2 * i], xi = qrow[DNOPE + 2 * i + 1];
    float c  = fcos[(size_t)s * (DROPE / 2) + i];
    float sn = fsin[(size_t)s * (DROPE / 2) + i];
    qv = ((d & 1) == 0) ? (xr * c - xi * sn) : (xr * sn + xi * c);
  }
  Qh[(bh * S_ + s) * QKD + d] = (__bf16)qv;

  float kv;
  if (d < DNOPE) kv = kvb[(size_t)row * (H_ * (DNOPE + DV)) + (size_t)h * (DNOPE + DV) + d];
  else           kv = kpe[(size_t)row * DROPE + (d - DNOPE)];
  Kh[(bh * S_ + s) * QKD + d] = (__bf16)kv;

  if (d < DV) {
    float vv = kvb[(size_t)row * (H_ * (DNOPE + DV)) + (size_t)h * (DNOPE + DV) + DNOPE + d];
    Vt[(bh * DV + d) * S_ + s] = (__bf16)vv;   // transposed: (128, S)
  }
}

// -------- generic bf16 WMMA GEMM: C[M,N] = A[M,K] * B[N,K]^T --------
// Each wave owns a 16x64 output strip: A fragment loaded once per k-step and
// reused across 4 B tiles (4x lower A/L2 traffic, 4 WMMAs per 10 b128 loads).
#define GEMM_WPB 4
#define GEMM_NT  4
__global__ __launch_bounds__(32 * GEMM_WPB)
void k_gemm_bf16(const __bf16* __restrict__ A, const __bf16* __restrict__ Bm,
                 float* __restrict__ C, int M, int N, int K) {
  int wave = threadIdx.x >> 5;
  int lane = threadIdx.x & 31;
  int tm = blockIdx.x;
  int n0 = (blockIdx.y * GEMM_WPB + wave) * (16 * GEMM_NT);
  if (n0 >= N) return;                 // wave-uniform; EXEC all-ones inside
  int rlo = lane & 15;
  int hi  = lane >> 4;
  const __bf16* Arow = A + (size_t)(tm * 16 + rlo) * K + hi * 16;
  const __bf16* Brow[GEMM_NT];
#pragma unroll
  for (int i = 0; i < GEMM_NT; ++i)
    Brow[i] = Bm + (size_t)(n0 + i * 16 + rlo) * K + hi * 16;

  const v8f vzero = {};
  v8f acc[GEMM_NT];
#pragma unroll
  for (int i = 0; i < GEMM_NT; ++i) acc[i] = vzero;

  for (int k0 = 0; k0 < K; k0 += 32) {
    v16bf a = load_a_frag(Arow + k0);
    __builtin_prefetch(Arow + k0 + 256, 0, 1);       // global_prefetch_b8
#pragma unroll
    for (int i = 0; i < GEMM_NT; ++i) {
      v16bf b = *(const v16bf*)(Brow[i] + k0);       // contiguous 32B per lane
      __builtin_prefetch(Brow[i] + k0 + 256, 0, 1);
      acc[i] = wmma_bf16(a, b, acc[i]);
    }
  }
#pragma unroll
  for (int i = 0; i < GEMM_NT; ++i) {
    float* Cout = C + (size_t)(tm * 16 + hi * 8) * N + n0 + i * 16 + rlo;
#pragma unroll
    for (int v = 0; v < 8; ++v) Cout[(size_t)v * N] = acc[i][v];
  }
}

// ---------------- flash attention, one wave per 16-query tile ----------------
__global__ __launch_bounds__(32)
void k_attn(const __bf16* __restrict__ Qh, const __bf16* __restrict__ Kh,
            const __bf16* __restrict__ Vt, __bf16* __restrict__ O) {
  const float scale = 0.07216878364870323f;   // 1/sqrt(192)
  const float NEG   = -3.0e38f;
  int lane = threadIdx.x;
  int qt = blockIdx.x;                 // S/16 tiles
  int bh = blockIdx.y;                 // b*H + h
  int b = bh / H_, h = bh % H_;
  const __bf16* Q = Qh + (size_t)bh * S_ * QKD;
  const __bf16* K = Kh + (size_t)bh * S_ * QKD;
  const __bf16* V = Vt + (size_t)bh * DV * S_;
  int rlo = lane & 15, hi = lane >> 4;

  // Preload 6 A-fragments of Q (covers QKD=192 = 6*32)
  const __bf16* Qrow = Q + (size_t)(qt * 16 + rlo) * QKD + hi * 16;
  v16bf qa[6];
#pragma unroll
  for (int kk = 0; kk < 6; ++kk) qa[kk] = load_a_frag(Qrow + kk * 32);

  const v8f vzero = {};
  v8f acc[8];                           // 16x128 output tile
#pragma unroll
  for (int nt = 0; nt < 8; ++nt) acc[nt] = vzero;
  float rmax[8], rsum[8];
#pragma unroll
  for (int v = 0; v < 8; ++v) { rmax[v] = NEG; rsum[v] = 0.f; }

  __shared__ __bf16 Pl[16 * 32];        // P tile staged for layout conversion
  int tmax = qt * 16 + 15;              // causal: last key this tile can see

  for (int t0 = 0; t0 <= tmax; t0 += 32) {
    // ---- S = Q * K^T for 32 keys (two 16-key N-tiles) ----
    v8f s0 = vzero, s1 = vzero;
    const __bf16* Kr0 = K + (size_t)(t0 + rlo) * QKD + hi * 16;
    const __bf16* Kr1 = Kr0 + (size_t)16 * QKD;
#pragma unroll
    for (int kk = 0; kk < 6; ++kk) {
      v16bf b0 = *(const v16bf*)(Kr0 + kk * 32);
      s0 = wmma_bf16(qa[kk], b0, s0);
      v16bf b1 = *(const v16bf*)(Kr1 + kk * 32);
      s1 = wmma_bf16(qa[kk], b1, s1);
    }
    // ---- online softmax (per C-layout row: element v, 16-lane half) ----
    int key0 = t0 + rlo, key1 = key0 + 16;
#pragma unroll
    for (int v = 0; v < 8; ++v) {
      int qidx = qt * 16 + hi * 8 + v;
      float sv0 = s0[v] * scale + ((key0 > qidx) ? NEG : 0.f);
      float sv1 = s1[v] * scale + ((key1 > qidx) ? NEG : 0.f);
      float mc = fmaxf(sv0, sv1);
#pragma unroll
      for (int off = 1; off < 16; off <<= 1) mc = fmaxf(mc, __shfl_xor(mc, off, 16));
      float mnew  = fmaxf(rmax[v], mc);
      float alpha = __expf(rmax[v] - mnew);
      float p0 = __expf(sv0 - mnew);
      float p1 = __expf(sv1 - mnew);
      float ps = p0 + p1;
#pragma unroll
      for (int off = 1; off < 16; off <<= 1) ps += __shfl_xor(ps, off, 16);
      rsum[v] = rsum[v] * alpha + ps;
      rmax[v] = mnew;
#pragma unroll
      for (int nt = 0; nt < 8; ++nt) acc[nt][v] *= alpha;
      Pl[(hi * 8 + v) * 32 + rlo]      = (__bf16)p0;
      Pl[(hi * 8 + v) * 32 + 16 + rlo] = (__bf16)p1;
    }
    __syncthreads();                    // single-wave block: ds wait
    // ---- reload P as A-fragment (16x32), accumulate P*V ----
    v16bf pa = load_a_frag(&Pl[rlo * 32 + hi * 16]);
    __syncthreads();
#pragma unroll
    for (int nt = 0; nt < 8; ++nt) {
      const __bf16* Vr = V + (size_t)(nt * 16 + rlo) * S_ + t0 + hi * 16;
      v16bf bv = *(const v16bf*)(Vr);   // contiguous thanks to V^T layout
      acc[nt] = wmma_bf16(pa, bv, acc[nt]);
    }
  }

  // ---- normalize + store O (row-major (R, H*DV) bf16) ----
  size_t orow_base = (size_t)(b * S_ + qt * 16 + hi * 8) * (H_ * DV);
#pragma unroll
  for (int v = 0; v < 8; ++v) {
    float inv = 1.0f / rsum[v];
    size_t orow = orow_base + (size_t)v * (H_ * DV);
#pragma unroll
    for (int nt = 0; nt < 8; ++nt)
      O[orow + h * DV + nt * 16 + rlo] = (__bf16)(acc[nt][v] * inv);
  }
}

// =============================== host launcher ===============================
extern "C" void kernel_launch(void* const* d_in, const int* in_sizes, int n_in,
                              void* d_out, int out_size, void* d_ws, size_t ws_size,
                              hipStream_t stream) {
  (void)in_sizes; (void)n_in; (void)out_size; (void)ws_size;
  const float* x         = (const float*)d_in[0];
  const float* fcos      = (const float*)d_in[1];
  const float* fsin      = (const float*)d_in[2];
  /* d_in[3] = mask: causal, computed inline in k_attn */
  const float* wq_a      = (const float*)d_in[4];
  const float* q_norm_w  = (const float*)d_in[5];
  const float* wq_b      = (const float*)d_in[6];
  const float* wkv_a     = (const float*)d_in[7];
  const float* kv_norm_w = (const float*)d_in[8];
  const float* wkv_b     = (const float*)d_in[9];
  const float* wo        = (const float*)d_in[10];
  float* out = (float*)d_out;

  char* ws = (char*)d_ws;
  size_t off = 0;
  auto carve = [&](size_t elems, size_t esz) -> char* {
    char* p = ws + off;
    off += (elems * esz + 255) & ~(size_t)255;
    return p;
  };
  __bf16* xb     = (__bf16*)carve((size_t)R_ * D_, 2);
  __bf16* wqa_b  = (__bf16*)carve((size_t)QLORA * D_, 2);
  __bf16* wqb_b  = (__bf16*)carve((size_t)H_ * QKD * QLORA, 2);
  __bf16* wkva_b = (__bf16*)carve((size_t)(KVLORA + DROPE) * D_, 2);
  __bf16* wkvb_b = (__bf16*)carve((size_t)H_ * (DNOPE + DV) * KVLORA, 2);
  __bf16* wo_b   = (__bf16*)carve((size_t)D_ * H_ * DV, 2);
  float*  qa_f   = (float*) carve((size_t)R_ * QLORA, 4);
  __bf16* qn_b   = (__bf16*)carve((size_t)R_ * QLORA, 2);
  float*  q_f    = (float*) carve((size_t)R_ * H_ * QKD, 4);
  float*  kvf_f  = (float*) carve((size_t)R_ * (KVLORA + DROPE), 4);
  __bf16* kvn_b  = (__bf16*)carve((size_t)R_ * KVLORA, 2);
  float*  kpe_f  = (float*) carve((size_t)R_ * DROPE, 4);
  float*  kvb_f  = (float*) carve((size_t)R_ * H_ * (DNOPE + DV), 4);
  __bf16* Qh     = (__bf16*)carve((size_t)B_ * H_ * S_ * QKD, 2);
  __bf16* Kh     = (__bf16*)carve((size_t)B_ * H_ * S_ * QKD, 2);
  __bf16* Vt     = (__bf16*)carve((size_t)B_ * H_ * DV * S_, 2);
  __bf16* O_b    = (__bf16*)carve((size_t)R_ * H_ * DV, 2);

  // fp32 -> bf16 casts (activations + weights), 4 elems/thread
  k_f32_to_bf16<<<2048, 256, 0, stream>>>(x, xb, (size_t)R_ * D_ / 4);
  k_f32_to_bf16<<<2048, 256, 0, stream>>>(wq_a, wqa_b, (size_t)QLORA * D_ / 4);
  k_f32_to_bf16<<<2048, 256, 0, stream>>>(wq_b, wqb_b, (size_t)H_ * QKD * QLORA / 4);
  k_f32_to_bf16<<<2048, 256, 0, stream>>>(wkv_a, wkva_b, (size_t)(KVLORA + DROPE) * D_ / 4);
  k_f32_to_bf16<<<2048, 256, 0, stream>>>(wkv_b, wkvb_b, (size_t)H_ * (DNOPE + DV) * KVLORA / 4);
  k_f32_to_bf16<<<2048, 256, 0, stream>>>(wo, wo_b, (size_t)D_ * H_ * DV / 4);

  // q chain: qa = x @ wq_a^T ; rmsnorm ; q = qn @ wq_b^T
  k_gemm_bf16<<<dim3(R_ / 16, QLORA / 256), 32 * GEMM_WPB, 0, stream>>>(
      xb, wqa_b, qa_f, R_, QLORA, D_);
  k_rmsnorm<<<R_, 256, 0, stream>>>(qa_f, q_norm_w, qn_b, QLORA);
  k_gemm_bf16<<<dim3(R_ / 16, (H_ * QKD) / 256), 32 * GEMM_WPB, 0, stream>>>(
      qn_b, wqb_b, q_f, R_, H_ * QKD, QLORA);

  // kv chain: kvf = x @ wkv_a^T ; split (rmsnorm + rope) ; kvb = kvn @ wkv_b^T
  k_gemm_bf16<<<dim3(R_ / 16, ((KVLORA + DROPE) + 255) / 256), 32 * GEMM_WPB, 0, stream>>>(
      xb, wkva_b, kvf_f, R_, KVLORA + DROPE, D_);
  k_kv_split<<<R_, 256, 0, stream>>>(kvf_f, kv_norm_w, fcos, fsin, kvn_b, kpe_f);
  k_gemm_bf16<<<dim3(R_ / 16, (H_ * (DNOPE + DV)) / 256), 32 * GEMM_WPB, 0, stream>>>(
      kvn_b, wkvb_b, kvb_f, R_, H_ * (DNOPE + DV), KVLORA);

  // assemble per-head Q/K/V^T (rope on q_pe), run flash attention
  k_assemble<<<dim3(R_, H_), QKD, 0, stream>>>(q_f, kvb_f, kpe_f, fcos, fsin, Qh, Kh, Vt);
  k_attn<<<dim3(S_ / 16, B_ * H_), 32, 0, stream>>>(Qh, Kh, Vt, O_b);

  // out = O @ wo^T (fp32 out)
  k_gemm_bf16<<<dim3(R_ / 16, D_ / 256), 32 * GEMM_WPB, 0, stream>>>(
      O_b, wo_b, out, R_, D_, H_ * DV);
}
